// ForestPointNetPP_79534204387678
// MI455X (gfx1250) — matrix operand order repaired and checked by reading, not compiled, not run on hardware
//
#include <hip/hip_runtime.h>
#include <hip/hip_bf16.h>

// ---- CDNA5 WMMA types ----
typedef __attribute__((ext_vector_type(16))) __bf16 v16bf;
typedef __attribute__((ext_vector_type(8)))  __bf16 v8bf;
typedef __attribute__((ext_vector_type(8)))  float  v8f;
typedef __attribute__((ext_vector_type(4)))  int    v4i;

#define BN_SCALE_F 0.9999950000374997f   // 1/sqrt(1+1e-5)
#define FLT_BIG    3.402823466e38f

// CDNA5 async global->LDS copy path (ASYNCcnt). Probe-confirmed present on
// this toolchain; signature wants AS1 v4i* src, AS3 v4i* dst, imm offset, cpol.
#if __has_builtin(__builtin_amdgcn_global_load_async_to_lds_b128) && \
    __has_builtin(__builtin_amdgcn_s_wait_asynccnt)
#define USE_ASYNC_LDS 1
typedef __attribute__((address_space(1))) v4i gv4i;
typedef __attribute__((address_space(3))) v4i lv4i;
#endif

// =====================================================================
// Weight packing: W f32 [Cin,Cout] -> Wt bf16 [Npad,Kpad] (row = out ch,
// contiguous K) + padded bias. Zero padding so WMMA K/N tiles are exact.
// =====================================================================
__global__ void pack_wt_kernel(const float* __restrict__ W, const float* __restrict__ b,
                               __bf16* __restrict__ Wt, float* __restrict__ bp,
                               int Cin, int Cout, int Kpad, int Npad) {
    int gid = blockIdx.x * blockDim.x + threadIdx.x;
    int total = Npad * Kpad;
    if (gid < Npad) bp[gid] = (gid < Cout) ? b[gid] : 0.0f;
    if (gid >= total) return;
    int n = gid / Kpad, k = gid % Kpad;
    float v = (n < Cout && k < Cin) ? W[(size_t)k * Cout + n] : 0.0f;
    Wt[gid] = (__bf16)v;
}

// =====================================================================
// Farthest point sampling: ONE workgroup, whole point cloud LDS-resident.
// MI455X: 320KB LDS/WGP -> 16384 pts (192KB pos + 64KB dmin) fits.
// =====================================================================
__global__ __launch_bounds__(1024) void fps_kernel(const float* __restrict__ pos,
                                                   int N, int M,
                                                   int* __restrict__ idx,
                                                   float* __restrict__ cpos) {
    extern __shared__ char smem[];
    float* sp   = (float*)smem;          // N*3
    float* dmin = sp + (size_t)3 * N;    // N
    float* rv   = dmin + N;              // blockDim
    int*   ri   = (int*)(rv + blockDim.x);
    __shared__ int s_last;
    int tid = threadIdx.x;
    for (int i = tid; i < N; i += blockDim.x) {
        sp[3*i+0] = pos[3*i+0]; sp[3*i+1] = pos[3*i+1]; sp[3*i+2] = pos[3*i+2];
        dmin[i] = FLT_BIG;
    }
    if (tid == 0) s_last = 0;
    __syncthreads();
    for (int it = 0; it < M; ++it) {
        int last = s_last;
        if (tid == 0) {
            idx[it] = last;
            cpos[3*it+0] = sp[3*last+0]; cpos[3*it+1] = sp[3*last+1]; cpos[3*it+2] = sp[3*last+2];
        }
        float lx = sp[3*last+0], ly = sp[3*last+1], lz = sp[3*last+2];
        float bestd = -1.0f; int besti = 0;
        for (int i = tid; i < N; i += blockDim.x) {
            float dx = sp[3*i+0]-lx, dy = sp[3*i+1]-ly, dz = sp[3*i+2]-lz;
            float d = dx*dx + dy*dy + dz*dz;
            float dm = fminf(dmin[i], d);
            dmin[i] = dm;
            if (dm > bestd) { bestd = dm; besti = i; }
        }
        rv[tid] = bestd; ri[tid] = besti;
        __syncthreads();
        for (int s = blockDim.x >> 1; s > 0; s >>= 1) {
            if (tid < s && rv[tid+s] > rv[tid]) { rv[tid] = rv[tid+s]; ri[tid] = ri[tid+s]; }
            __syncthreads();
        }
        if (tid == 0) s_last = ri[0];
        __syncthreads();
    }
}

// =====================================================================
// 128-nearest selection per center; per-center distance array in LDS.
// =====================================================================
__global__ __launch_bounds__(128) void knn_kernel(const float* __restrict__ cpos,
                                                  const float* __restrict__ pos, int N,
                                                  int* __restrict__ nbr, float* __restrict__ nd2) {
    extern __shared__ char smem[];
    float* d2 = (float*)smem;     // N
    float* rv = d2 + N;           // 128
    int*   ri = (int*)(rv + 128); // 128
    int c = blockIdx.x, tid = threadIdx.x;
    float cx = cpos[3*c+0], cy = cpos[3*c+1], cz = cpos[3*c+2];
    for (int i = tid; i < N; i += 128) {
        float dx = pos[3*i+0]-cx, dy = pos[3*i+1]-cy, dz = pos[3*i+2]-cz;
        d2[i] = dx*dx + dy*dy + dz*dz;
    }
    __syncthreads();
    for (int r = 0; r < 128; ++r) {
        float best = FLT_BIG; int bi = 0;
        for (int i = tid; i < N; i += 128) {
            float v = d2[i];
            if (v < best) { best = v; bi = i; }
        }
        rv[tid] = best; ri[tid] = bi;
        __syncthreads();
        for (int s = 64; s > 0; s >>= 1) {
            if (tid < s && rv[tid+s] < rv[tid]) { rv[tid] = rv[tid+s]; ri[tid] = ri[tid+s]; }
            __syncthreads();
        }
        if (tid == 0) {
            nbr[(size_t)c*128 + r] = ri[0];
            nd2[(size_t)c*128 + r] = rv[0];
            d2[ri[0]] = FLT_BIG;
        }
        __syncthreads();
    }
}

// =====================================================================
// Gather neighbor features + relative position into padded bf16 rows.
// =====================================================================
__global__ void gather_kernel(const float* __restrict__ xprev, int F,
                              const float* __restrict__ ppos, const float* __restrict__ cpos,
                              const int* __restrict__ nbr, int C,
                              __bf16* __restrict__ feat, int Kpad) {
    size_t gid = (size_t)blockIdx.x * blockDim.x + threadIdx.x;
    size_t total = (size_t)C * 128 * Kpad;
    if (gid >= total) return;
    size_t row = gid / Kpad; int f = (int)(gid % Kpad);
    int c = (int)(row >> 7);
    int j = nbr[row];
    float v = 0.0f;
    if (f < F)            v = xprev[(size_t)j * F + f];
    else if (f < F + 3)   v = ppos[3*j + (f-F)] - cpos[3*c + (f-F)];
    feat[gid] = (__bf16)v;
}

// =====================================================================
// WMMA GEMM: Y[M,Npad] = act(X[M,K] @ Wt^T + b).
// Block = 8 waves = 8 M-tiles x 1 N-tile. The block's 16-row weight tile is
// staged once into LDS (async global->LDS on CDNA5, ASYNCcnt tracked) and
// reused by all 8 waves; A fragments stream from global with near-cache
// prefetch. bf16 A/B, fp32 accumulate via v_wmma_f32_16x16x32_bf16.
// mode: 0=none, 1=relu, 2=relu*BN_SCALE.  outf32: fp32 vs bf16 output.
// Fragment layouts per CDNA5 ISA 7.12.2 (wave32).
// =====================================================================
__global__ __launch_bounds__(256) void gemm_bf16_kernel(const __bf16* __restrict__ X,
                                                        const __bf16* __restrict__ Wt,
                                                        const float* __restrict__ bias,
                                                        int M, int K, int Npad,
                                                        void* __restrict__ Y, int Ystride,
                                                        int mode, int outf32) {
    __shared__ __bf16 sW[16 * 384];    // 16 output channels x K (K <= 384) = 12KB

    const int lane = threadIdx.x & 31;
    const int wave = threadIdx.x >> 5;
    const int half = lane >> 4;        // 0: lanes 0-15, 1: lanes 16-31
    const int l15  = lane & 15;
    const int mGroups = M >> 7;        // 8 tiles of 16 rows per block
    const int bm = blockIdx.x % mGroups;
    const int tn = blockIdx.x / mGroups;

    // ---- stage 16 x K weight tile into LDS (16B chunks per lane) ----
    {
        const int chunksPerRow = K >> 3;           // 8 bf16 = 16B per chunk
        const int chunks = 16 * chunksPerRow;
        for (int ci = threadIdx.x; ci < chunks; ci += blockDim.x) {
            int r  = ci / chunksPerRow;
            int kk = (ci % chunksPerRow) << 3;
            const __bf16* g = Wt + (size_t)(tn * 16 + r) * K + kk;
            __bf16*       l = &sW[r * K + kk];
#ifdef USE_ASYNC_LDS
            __builtin_amdgcn_global_load_async_to_lds_b128(
                (gv4i*)g, (lv4i*)l, 0, 0);
#else
            *(v8bf*)l = *(const v8bf*)g;
#endif
        }
#ifdef USE_ASYNC_LDS
        __builtin_amdgcn_s_wait_asynccnt(0);
#endif
        __syncthreads();
    }

    const int tm = bm * 8 + wave;
    const int m  = tm * 16 + l15;      // A row (both lane halves cover M=0..15)
    const int n  = tn * 16 + l15;      // B column / output column
    v8f acc = {};
    const __bf16* xbase = X + (size_t)m * K;
    const __bf16* wlds  = &sW[l15 * K];
    for (int k0 = 0; k0 < K; k0 += 32) {
        // A 16x32 bf16 fragment: lanes 0-15 hold K {0..7,16..23}, lanes 16-31 {8..15,24..31}
        const __bf16* xr = xbase + k0 + (half ? 8 : 0);
        v8bf alo = *(const v8bf*)(xr);
        v8bf ahi = *(const v8bf*)(xr + 16);
        if (k0 + 32 < K)
            __builtin_prefetch((const void*)(xr + 32), 0, 3);   // global_prefetch_b8, near
        // B 32x16 bf16 fragment from LDS: lanes 0-15 K {0..7,16..23}, lanes 16-31 {8..15,24..31}
        const __bf16* wr = wlds + k0 + (half ? 8 : 0);
        v8bf blo = *(const v8bf*)(wr);
        v8bf bhi = *(const v8bf*)(wr + 16);
        v16bf a = __builtin_shufflevector(alo, ahi, 0,1,2,3,4,5,6,7,8,9,10,11,12,13,14,15);
        v16bf b = __builtin_shufflevector(blo, bhi, 0,1,2,3,4,5,6,7,8,9,10,11,12,13,14,15);
        acc = __builtin_amdgcn_wmma_f32_16x16x32_bf16(false, a, false, b,
                                                      (short)0, acc, false, false);
    }
    const float bb = bias[n];
    #pragma unroll
    for (int r = 0; r < 8; ++r) {
        // C/D layout: VGPR r -> M = r (+8 for lanes 16-31), N = lane&15
        int mo = tm * 16 + r + half * 8;
        float v = acc[r] + bb;
        if (mode >= 1) v = fmaxf(v, 0.0f);
        if (mode == 2) v *= BN_SCALE_F;
        if (outf32) ((float*)Y)[(size_t)mo * Ystride + n] = v;
        else        ((__bf16*)Y)[(size_t)mo * Ystride + n] = (__bf16)v;
    }
}

// =====================================================================
// Radius-masked max aggregation over 128 neighbors.
// =====================================================================
__global__ void masked_max_kernel(const float* __restrict__ msg, const float* __restrict__ nd2,
                                  int C, int F, float r2,
                                  float* __restrict__ xo, int xstride, int off) {
    int gid = blockIdx.x * blockDim.x + threadIdx.x;
    if (gid >= C * F) return;
    int c = gid / F, f = gid % F;
    float best = -FLT_BIG;
    const float*  mrow = msg + (size_t)c * 128 * F + f;
    const float*  drow = nd2 + (size_t)c * 128;
    for (int k = 0; k < 128; ++k) {
        if (drow[k] <= r2) best = fmaxf(best, mrow[(size_t)k * F]);
    }
    xo[(size_t)c * xstride + off + f] = best;
}

// =====================================================================
// 3-NN inverse-distance interpolation.
// =====================================================================
__global__ void knn3_kernel(const float* __restrict__ pf, int Nf,
                            const float* __restrict__ pc, int Nc,
                            const float* __restrict__ xc, int F,
                            float* __restrict__ out) {
    int i = blockIdx.x * blockDim.x + threadIdx.x;
    if (i >= Nf) return;
    float px = pf[3*i+0], py = pf[3*i+1], pz = pf[3*i+2];
    float d0 = FLT_BIG, d1 = FLT_BIG, d2v = FLT_BIG;
    int i0 = 0, i1 = 0, i2 = 0;
    for (int j = 0; j < Nc; ++j) {
        float dx = pc[3*j+0]-px, dy = pc[3*j+1]-py, dz = pc[3*j+2]-pz;
        float d = dx*dx + dy*dy + dz*dz;
        if (d < d0)      { d2v=d1; i2=i1; d1=d0; i1=i0; d0=d; i0=j; }
        else if (d < d1) { d2v=d1; i2=i1; d1=d;  i1=j; }
        else if (d < d2v){ d2v=d;  i2=j; }
    }
    float w0 = 1.0f / fmaxf(d0, 1e-16f);
    float w1 = 1.0f / fmaxf(d1, 1e-16f);
    float w2 = 1.0f / fmaxf(d2v, 1e-16f);
    float ws = w0 + w1 + w2;
    const float* r0 = xc + (size_t)i0 * F;
    const float* r1 = xc + (size_t)i1 * F;
    const float* r2 = xc + (size_t)i2 * F;
    float* o = out + (size_t)i * F;
    for (int f = 0; f < F; ++f)
        o[f] = (w0*r0[f] + w1*r1[f] + w2*r2[f]) / ws;
}

// =====================================================================
// Concat two fp32 feature blocks into a padded bf16 matrix.
// =====================================================================
__global__ void concat_kernel(const float* __restrict__ A, int Fa,
                              const float* __restrict__ B, int Fb,
                              int M, __bf16* __restrict__ out, int Kpad) {
    size_t gid = (size_t)blockIdx.x * blockDim.x + threadIdx.x;
    size_t total = (size_t)M * Kpad;
    if (gid >= total) return;
    size_t row = gid / Kpad; int f = (int)(gid % Kpad);
    float v = 0.0f;
    if (f < Fa)           v = A[row * Fa + f];
    else if (f < Fa + Fb) v = B[row * Fb + (f - Fa)];
    out[gid] = (__bf16)v;
}

// =====================================================================
// 2-class log-softmax.
// =====================================================================
__global__ void logsm_kernel(const float* __restrict__ logits, int N, float* __restrict__ out) {
    int i = blockIdx.x * blockDim.x + threadIdx.x;
    if (i >= N) return;
    float a = logits[(size_t)i*16 + 0], b = logits[(size_t)i*16 + 1];
    float m = fmaxf(a, b);
    float lse = m + logf(expf(a - m) + expf(b - m));
    out[2*i+0] = a - lse;
    out[2*i+1] = b - lse;
}

// =====================================================================
// Host orchestration
// =====================================================================
extern "C" void kernel_launch(void* const* d_in, const int* in_sizes, int n_in,
                              void* d_out, int out_size, void* d_ws, size_t ws_size,
                              hipStream_t stream) {
    (void)in_sizes; (void)n_in; (void)out_size; (void)ws_size;
    const float* x   = (const float*)d_in[0];
    const float* pos = (const float*)d_in[1];
    const int N = 16384, C1 = 4096, C2 = 1024, C3 = 256, K = 128;

    // Layers, d_in indices assuming dict insertion order:
    // x, pos, sa1(m0 3L, m1 3L), sa2, sa3, fp3(2L), fp2(2L), fp1(3L), clf(3L)
    struct LD { int wi, Cin, Cout; };
    static const LD L[28] = {
        { 2,  8, 32},{ 4, 32, 32},{ 6, 32, 32},   //  0.. 2  sa1 m0
        { 8,  8, 32},{10, 32, 32},{12, 32, 32},   //  3.. 5  sa1 m1
        {14, 67, 64},{16, 64, 64},{18, 64, 64},   //  6.. 8  sa2 m0
        {20, 67, 64},{22, 64, 64},{24, 64, 64},   //  9..11  sa2 m1
        {26,131,128},{28,128,128},{30,128,128},   // 12..14  sa3 m0
        {32,131,128},{34,128,128},{36,128,128},   // 15..17  sa3 m1
        {38,384,256},{40,256,256},                // 18..19  fp3
        {42,320,256},{44,256,128},                // 20..21  fp2
        {46,133,128},{48,128,128},{50,128,128},   // 22..24  fp1
        {52,128,128},{54,128, 64},{56, 64,  2},   // 25..27  clf
    };

    char* base = (char*)d_ws;
    size_t off = 0;
    auto alloc = [&](size_t bytes) -> void* {
        void* p = base + off;
        off = (off + bytes + 255) & ~(size_t)255;
        return p;
    };

    int Kp[28], Np[28];
    __bf16* Wt[28]; float* Bp[28];
    for (int i = 0; i < 28; ++i) {
        Kp[i] = (L[i].Cin + 31) / 32 * 32;
        Np[i] = (L[i].Cout + 15) / 16 * 16;
        Wt[i] = (__bf16*)alloc((size_t)Np[i] * Kp[i] * sizeof(__bf16));
        Bp[i] = (float*)alloc((size_t)Np[i] * sizeof(float));
    }
    for (int i = 0; i < 28; ++i) {
        int elems = Np[i] * Kp[i];
        pack_wt_kernel<<<(elems + 255) / 256, 256, 0, stream>>>(
            (const float*)d_in[L[i].wi], (const float*)d_in[L[i].wi + 1],
            Wt[i], Bp[i], L[i].Cin, L[i].Cout, Kp[i], Np[i]);
    }

    // scratch buffers
    int*   idx1 = (int*)alloc(C1*4);           float* cp1 = (float*)alloc((size_t)C1*12);
    int*   idx2 = (int*)alloc(C2*4);           float* cp2 = (float*)alloc((size_t)C2*12);
    int*   idx3 = (int*)alloc(C3*4);           float* cp3 = (float*)alloc((size_t)C3*12);
    int*   nbr1 = (int*)alloc((size_t)C1*K*4); float* nds1 = (float*)alloc((size_t)C1*K*4);
    int*   nbr2 = (int*)alloc((size_t)C2*K*4); float* nds2 = (float*)alloc((size_t)C2*K*4);
    int*   nbr3 = (int*)alloc((size_t)C3*K*4); float* nds3 = (float*)alloc((size_t)C3*K*4);
    __bf16* feat1 = (__bf16*)alloc((size_t)C1*K*32*2);
    __bf16* hA1   = (__bf16*)alloc((size_t)C1*K*32*2);
    __bf16* hB1   = (__bf16*)alloc((size_t)C1*K*32*2);
    float*  msg1  = (float*) alloc((size_t)C1*K*32*4);
    float*  x1    = (float*) alloc((size_t)C1*64*4);
    __bf16* feat2 = (__bf16*)alloc((size_t)C2*K*96*2);
    __bf16* hA2   = (__bf16*)alloc((size_t)C2*K*64*2);
    __bf16* hB2   = (__bf16*)alloc((size_t)C2*K*64*2);
    float*  msg2  = (float*) alloc((size_t)C2*K*64*4);
    float*  x2    = (float*) alloc((size_t)C2*128*4);
    __bf16* feat3 = (__bf16*)alloc((size_t)C3*K*160*2);
    __bf16* hA3   = (__bf16*)alloc((size_t)C3*K*128*2);
    __bf16* hB3   = (__bf16*)alloc((size_t)C3*K*128*2);
    float*  msg3  = (float*) alloc((size_t)C3*K*128*4);
    float*  x3    = (float*) alloc((size_t)C3*256*4);
    float*  it3   = (float*) alloc((size_t)C2*256*4);
    __bf16* cat3  = (__bf16*)alloc((size_t)C2*384*2);
    __bf16* f3a   = (__bf16*)alloc((size_t)C2*256*2);
    float*  u3    = (float*) alloc((size_t)C2*256*4);
    float*  it2   = (float*) alloc((size_t)C1*256*4);
    __bf16* cat2  = (__bf16*)alloc((size_t)C1*320*2);
    __bf16* f2a   = (__bf16*)alloc((size_t)C1*256*2);
    float*  u2    = (float*) alloc((size_t)C1*128*4);
    float*  it1   = (float*) alloc((size_t)N*128*4);
    __bf16* cat1  = (__bf16*)alloc((size_t)N*160*2);
    __bf16* f1a   = (__bf16*)alloc((size_t)N*128*2);
    __bf16* f1b   = (__bf16*)alloc((size_t)N*128*2);
    __bf16* u1    = (__bf16*)alloc((size_t)N*128*2);
    __bf16* hc1   = (__bf16*)alloc((size_t)N*128*2);
    __bf16* hc2   = (__bf16*)alloc((size_t)N*64*2);
    float*  logits= (float*) alloc((size_t)N*16*4);

    auto gemm = [&](const __bf16* X, int M, int li, void* Y, int Ystride, int mode, int f32o) {
        int blocks = (M / 128) * (Np[li] / 16);   // all M here are multiples of 128
        gemm_bf16_kernel<<<blocks, 256, 0, stream>>>(
            X, Wt[li], Bp[li], M, Kp[li], Np[li], Y, Ystride, mode, f32o);
    };

    // ---------------- SA1 ----------------
    fps_kernel<<<1, 1024, (size_t)16*N + 8192, stream>>>(pos, N, C1, idx1, cp1);
    knn_kernel<<<C1, 128, (size_t)4*N + 2048, stream>>>(cp1, pos, N, nbr1, nds1);
    {
        size_t tot = (size_t)C1 * K * 32;
        gather_kernel<<<(unsigned)((tot + 255) / 256), 256, 0, stream>>>(x, 5, pos, cp1, nbr1, C1, feat1, 32);
    }
    const float r1[2] = {0.05f, 0.4f};
    for (int s = 0; s < 2; ++s) {
        gemm(feat1, C1*K, 0 + 3*s, hA1, 32, 2, 0);
        gemm(hA1,   C1*K, 1 + 3*s, hB1, 32, 2, 0);
        gemm(hB1,   C1*K, 2 + 3*s, msg1, 32, 2, 1);
        masked_max_kernel<<<(C1*32 + 255) / 256, 256, 0, stream>>>(
            msg1, nds1, C1, 32, r1[s]*r1[s], x1, 64, 32*s);
    }
    // ---------------- SA2 ----------------
    fps_kernel<<<1, 1024, (size_t)16*C1 + 8192, stream>>>(cp1, C1, C2, idx2, cp2);
    knn_kernel<<<C2, 128, (size_t)4*C1 + 2048, stream>>>(cp2, cp1, C1, nbr2, nds2);
    {
        size_t tot = (size_t)C2 * K * 96;
        gather_kernel<<<(unsigned)((tot + 255) / 256), 256, 0, stream>>>(x1, 64, cp1, cp2, nbr2, C2, feat2, 96);
    }
    const float r2[2] = {0.2f, 0.8f};
    for (int s = 0; s < 2; ++s) {
        gemm(feat2, C2*K, 6 + 3*s, hA2, 64, 2, 0);
        gemm(hA2,   C2*K, 7 + 3*s, hB2, 64, 2, 0);
        gemm(hB2,   C2*K, 8 + 3*s, msg2, 64, 2, 1);
        masked_max_kernel<<<(C2*64 + 255) / 256, 256, 0, stream>>>(
            msg2, nds2, C2, 64, r2[s]*r2[s], x2, 128, 64*s);
    }
    // ---------------- SA3 ----------------
    fps_kernel<<<1, 1024, (size_t)16*C2 + 8192, stream>>>(cp2, C2, C3, idx3, cp3);
    knn_kernel<<<C3, 128, (size_t)4*C2 + 2048, stream>>>(cp3, cp2, C2, nbr3, nds3);
    {
        size_t tot = (size_t)C3 * K * 160;
        gather_kernel<<<(unsigned)((tot + 255) / 256), 256, 0, stream>>>(x2, 128, cp2, cp3, nbr3, C3, feat3, 160);
    }
    const float r3[2] = {0.4f, 1.6f};
    for (int s = 0; s < 2; ++s) {
        gemm(feat3, C3*K, 12 + 3*s, hA3, 128, 2, 0);
        gemm(hA3,   C3*K, 13 + 3*s, hB3, 128, 2, 0);
        gemm(hB3,   C3*K, 14 + 3*s, msg3, 128, 2, 1);
        masked_max_kernel<<<(C3*128 + 255) / 256, 256, 0, stream>>>(
            msg3, nds3, C3, 128, r3[s]*r3[s], x3, 256, 128*s);
    }
    // ---------------- FP3 ----------------
    knn3_kernel<<<(C2 + 255) / 256, 256, 0, stream>>>(cp2, C2, cp3, C3, x3, 256, it3);
    concat_kernel<<<(unsigned)(((size_t)C2*384 + 255) / 256), 256, 0, stream>>>(it3, 256, x2, 128, C2, cat3, 384);
    gemm(cat3, C2, 18, f3a, 256, 2, 0);
    gemm(f3a,  C2, 19, u3,  256, 2, 1);
    // ---------------- FP2 ----------------
    knn3_kernel<<<(C1 + 255) / 256, 256, 0, stream>>>(cp1, C1, cp2, C2, u3, 256, it2);
    concat_kernel<<<(unsigned)(((size_t)C1*320 + 255) / 256), 256, 0, stream>>>(it2, 256, x1, 64, C1, cat2, 320);
    gemm(cat2, C1, 20, f2a, 256, 2, 0);
    gemm(f2a,  C1, 21, u2,  128, 2, 1);
    // ---------------- FP1 ----------------
    knn3_kernel<<<(N + 255) / 256, 256, 0, stream>>>(pos, N, cp1, C1, u2, 128, it1);
    concat_kernel<<<(unsigned)(((size_t)N*160 + 255) / 256), 256, 0, stream>>>(it1, 128, x, 5, N, cat1, 160);
    gemm(cat1, N, 22, f1a, 128, 2, 0);
    gemm(f1a,  N, 23, f1b, 128, 2, 0);
    gemm(f1b,  N, 24, u1,  128, 2, 0);
    // ---------------- Classifier ----------------
    gemm(u1,  N, 25, hc1, 128, 1, 0);
    gemm(hc1, N, 26, hc2, 64,  1, 0);
    gemm(hc2, N, 27, logits, 16, 0, 1);
    logsm_kernel<<<(N + 255) / 256, 256, 0, stream>>>(logits, N, (float*)d_out);
}
